// GAT_71854802862669
// MI455X (gfx1250) — matrix operand order repaired and checked by reading, not compiled, not run on hardware
//
#include <hip/hip_runtime.h>
#include <hip/hip_bf16.h>

// ---------------------------------------------------------------------------
// GAT forward for MI455X (gfx1250, wave32, WMMA).
//   1. pack/convert weights + x to bf16; bit-pack adjacency (wave32 ballot)
//   2. Wh  = x @ W1packed                      (WMMA GEMM, 4096x512x512)
//   3. src/dst projections
//   4. per-row online-softmax stats (max,sum) streaming the 2MB adj bitmap
//   5. fused attention: P tile regenerated in regs from (src,dst,bits,stats),
//      P @ Wh via v_wmma_f32_16x16x32_bf16; 4 waves share one LDS Wh tile
//   6. layer-2 repeat with one 512-wide head
//   7. relu(x2 @ lin1_w + b) -> (@ lin2_w + b) -> d_out (f32)
// ---------------------------------------------------------------------------

typedef __attribute__((ext_vector_type(16))) __bf16 v16bf;
typedef __attribute__((ext_vector_type(8)))  float  v8f;

#define ALPHA_LRELU 0.2f
#define NEGV        (-9.0e15f)

// ---------------------------- small utility kernels ------------------------

__global__ void k_f32_to_bf16(const float* __restrict__ in,
                              __bf16* __restrict__ out, size_t cnt) {
  size_t i = (size_t)blockIdx.x * blockDim.x + threadIdx.x;
  size_t stride = (size_t)gridDim.x * blockDim.x;
  for (; i < cnt; i += stride) out[i] = (__bf16)in[i];
}

// W1 [H][F][O] f32  ->  W1b [F][H*O] bf16  (so one GEMM covers all heads)
__global__ void k_pack_W1(const float* __restrict__ W1, __bf16* __restrict__ out,
                          int H, int F, int O) {
  int idx = blockIdx.x * blockDim.x + threadIdx.x;
  int total = H * F * O;
  if (idx >= total) return;
  int h = idx / (F * O);
  int rem = idx % (F * O);
  int f = rem / O;
  int o = rem % O;
  out[(size_t)f * (H * O) + h * O + o] = (__bf16)W1[idx];
}

// adj [n][n] int32 0/1 -> bits [n][n/32] u32 via wave32 ballot (32x traffic cut)
__global__ __launch_bounds__(256) void k_pack_adj(const int* __restrict__ adj,
                                                  unsigned* __restrict__ bits,
                                                  int n) {
  int wave = threadIdx.x >> 5, lane = threadIdx.x & 31;
  int nw = n >> 5;
  int chunk = blockIdx.x * 8 + wave;  // over n*nw
  if (chunk >= n * nw) return;
  int i = chunk / nw, c = chunk % nw;
  int v = adj[(size_t)i * n + c * 32 + lane] != 0;
  unsigned long long b = __ballot(v);
  if (lane == 0) bits[chunk] = (unsigned)b;
}

// src[h,i] = Wh[i, h*O..]·a[h,0:O] ; dst[h,i] = Wh[i, h*O..]·a[h,O:2O]
__global__ void k_attn_proj(const float* __restrict__ Wh,
                            const float* __restrict__ a,
                            float* __restrict__ src, float* __restrict__ dst,
                            int n, int H, int O) {
  int idx = blockIdx.x * blockDim.x + threadIdx.x;
  if (idx >= H * n) return;
  int h = idx / n, i = idx % n;
  const float* row = Wh + (size_t)i * (H * O) + h * O;
  const float* av  = a + (size_t)h * 2 * O;
  float s = 0.f, d = 0.f;
  for (int o = 0; o < O; ++o) {
    float w = row[o];
    s += w * av[o];
    d += w * av[O + o];
  }
  src[idx] = s;
  dst[idx] = d;
}

// one wave per attention row: online softmax (max, sum) over masked
// leakyReLU(src_i + dst_j).  Each 32-j chunk uses ONE broadcast u32 adj word.
__global__ __launch_bounds__(256) void k_softmax_stats(
    const float* __restrict__ src, const float* __restrict__ dst,
    const unsigned* __restrict__ bits, float* __restrict__ mo,
    float* __restrict__ so, int n, int H) {
  int wave = threadIdx.x >> 5, lane = threadIdx.x & 31;
  int row = blockIdx.x * 8 + wave;  // row in [0, H*n)
  if (row >= H * n) return;
  int h = row / n, i = row % n;
  int nw = n >> 5;
  float si = src[row];
  const unsigned* brow = bits + (size_t)i * nw;
  const float*    drow = dst + (size_t)h * n;
  float m = -3.0e38f, s = 0.f;
  for (int j = lane; j < n; j += 32) {
    unsigned w = brow[j >> 5];  // same word across the wave (broadcast)
    float e = si + drow[j];
    e = e > 0.f ? e : ALPHA_LRELU * e;
    e = ((w >> (j & 31)) & 1u) ? e : NEGV;
    float nm = fmaxf(m, e);
    s = s * __expf(m - nm) + __expf(e - nm);
    m = nm;
  }
  for (int off = 16; off; off >>= 1) {
    float m2 = __shfl_xor(m, off, 32);
    float s2 = __shfl_xor(s, off, 32);
    float nm = fmaxf(m, m2);
    s = s * __expf(m - nm) + s2 * __expf(m2 - nm);
    m = nm;
  }
  if (lane == 0) { mo[row] = m; so[row] = s; }
}

// ------------------------- fused attention (P @ Wh) -------------------------
// 128 threads = 4 waves; wave w owns 16-row block (blockIdx.x*4 + w); all four
// waves share one LDS-staged 32x64 Wh tile per j step.  All four B fragments
// are preloaded before the WMMA burst so dscnt waits overlap the first WMMAs.
__global__ __launch_bounds__(128) void k_attn_wmma(
    const __bf16* __restrict__ Whb, const float* __restrict__ src,
    const float* __restrict__ dstv, const float* __restrict__ mrow,
    const float* __restrict__ srow, const unsigned* __restrict__ bits,
    float* __restrict__ outF, __bf16* __restrict__ outB,
    int n, int H, int O) {
  __shared__ __bf16 Wlds[64][40];  // [col][k] transposed, padded

  const int t = threadIdx.x;
  const int lane = t & 31;
  const int wave = t >> 5;
  const int cb = blockIdx.y;       // 64-col block
  const int h  = blockIdx.z;
  const int ldw = H * O;
  const int nw = n >> 5;
  const int colbase = h * O + cb * 64;

  const int rb = blockIdx.x * 4 + wave;  // this wave's 16-row block
  const int i = rb * 16 + (lane & 15);   // A-matrix row this lane feeds
  const int rowid = h * n + i;
  const float si  = src[rowid];
  const float mi  = mrow[rowid];
  const float inv = 1.f / srow[rowid];
  const unsigned* brow = bits + (size_t)i * nw;
  const float*    drow = dstv + (size_t)h * n;

  const int kb   = (lane < 16) ? 0 : 8;  // A-layout K base for this half-wave
  const int ks   = (lane >> 4) * 16;     // unused; B fragments also use kb
  const int nloc = lane & 15;

  v8f acc[4] = {};

  for (int j0 = 0; j0 < n; j0 += 32) {
    // cooperative stage of Wh[j0..j0+31, colbase..+63] (transposed into LDS):
    // thread t loads row j0+t/4, 16 contiguous bf16 at col (t%4)*16
    {
      int jr = t >> 2, c = (t & 3) * 16;
      const __bf16* wr = Whb + (size_t)(j0 + jr) * ldw + colbase + c;
#pragma unroll
      for (int q = 0; q < 16; ++q) Wlds[c + q][jr] = wr[q];
    }
    __syncthreads();

    // build A fragment = softmax probabilities; one u32 covers all 16 slots
    unsigned wbits = brow[j0 >> 5];
    v16bf afrag;
#pragma unroll
    for (int half = 0; half < 2; ++half) {
      int kofs = kb + half * 16;
      int jb = j0 + kofs;
#pragma unroll
      for (int q = 0; q < 8; ++q) {
        float e = si + drow[jb + q];
        e = e > 0.f ? e : ALPHA_LRELU * e;
        e = ((wbits >> (kofs + q)) & 1u) ? e : NEGV;
        afrag[half * 8 + q] = (__bf16)(__expf(e - mi) * inv);
      }
    }

    // preload all 4 B fragments, then burst the WMMAs
    v16bf bfrag[4];
#pragma unroll
    for (int nt = 0; nt < 4; ++nt)
#pragma unroll
      for (int q = 0; q < 16; ++q) bfrag[nt][q] = Wlds[nt * 16 + nloc][kb + q + (q >> 3) * 8];
#pragma unroll
    for (int nt = 0; nt < 4; ++nt)
      acc[nt] = __builtin_amdgcn_wmma_f32_16x16x32_bf16(
          false, afrag, false, bfrag[nt], (short)0, acc[nt], false, false);
    __syncthreads();
  }

  // epilogue: C layout VGPR r -> M = r + 8*(lane/16), N = lane%16 + 16*nt
  // apply ELU in place, then uniform-branch store loops (no per-store branch)
#pragma unroll
  for (int nt = 0; nt < 4; ++nt)
#pragma unroll
    for (int r = 0; r < 8; ++r) {
      float v = acc[nt][r];
      acc[nt][r] = v > 0.f ? v : (__expf(v) - 1.f);  // ELU
    }
  const int mbase = rb * 16 + (lane >> 4) * 8;
  if (outF) {
#pragma unroll
    for (int nt = 0; nt < 4; ++nt) {
      int col = colbase + nt * 16 + nloc;
#pragma unroll
      for (int r = 0; r < 8; ++r)
        outF[(size_t)(mbase + r) * ldw + col] = acc[nt][r];
    }
  }
  if (outB) {
#pragma unroll
    for (int nt = 0; nt < 4; ++nt) {
      int col = colbase + nt * 16 + nloc;
#pragma unroll
      for (int r = 0; r < 8; ++r)
        outB[(size_t)(mbase + r) * ldw + col] = (__bf16)acc[nt][r];
    }
  }
}

// ------------------------------ tiled WMMA GEMM -----------------------------
// C[M,N] (+bias, act) = A[M,K] @ B[K,N], all row-major, A/B bf16, C f32/bf16.
// Block = 256 threads = 8 waves; tile 128x64, K-step 32; each wave owns a
// 16x64 C stripe (4 accumulators).  B fragments preloaded before WMMA burst.
__global__ __launch_bounds__(256) void k_gemm_bf16(
    const __bf16* __restrict__ A, const __bf16* __restrict__ B,
    float* __restrict__ Cf, __bf16* __restrict__ Cb,
    const float* __restrict__ bias, int M, int N, int K, int act) {
  __shared__ __bf16 Alds[128][40];  // [m][k], padded
  __shared__ __bf16 Blds[64][40];   // [n][k] transposed, padded

  const int t = threadIdx.x;
  const int lane = t & 31;
  const int wave = t >> 5;
  const int m0 = blockIdx.x * 128;
  const int n0 = blockIdx.y * 64;

  v8f acc[4] = {};

  const int arow = wave * 16 + (lane & 15);
  const int kb   = (lane < 16) ? 0 : 8;
  const int ks   = (lane >> 4) * 16;
  const int nloc = lane & 15;

  for (int k0 = 0; k0 < K; k0 += 32) {
    // A tile: thread t -> row t/2, 16 bf16 at col (t&1)*16 (two b128 loads)
    {
      int r = t >> 1, c = (t & 1) * 16;
      const uint4* sp = (const uint4*)(A + (size_t)(m0 + r) * K + k0 + c);
      uint4* dp = (uint4*)(&Alds[r][c]);
      dp[0] = sp[0];
      dp[1] = sp[1];
    }
    // B tile transposed: thread t -> row k0+t/8, 8 bf16 at col n0+(t&7)*8
    {
      int r = t >> 3, c = (t & 7) * 8;
      const __bf16* sp = B + (size_t)(k0 + r) * N + n0 + c;
#pragma unroll
      for (int q = 0; q < 8; ++q) Blds[c + q][r] = sp[q];
    }
    __syncthreads();

    v16bf afrag;
#pragma unroll
    for (int q = 0; q < 8; ++q) {
      afrag[q]     = Alds[arow][kb + q];
      afrag[8 + q] = Alds[arow][kb + 16 + q];
    }
    v16bf bfrag[4];
#pragma unroll
    for (int nt = 0; nt < 4; ++nt)
#pragma unroll
      for (int q = 0; q < 16; ++q) bfrag[nt][q] = Blds[nt * 16 + nloc][kb + q + (q >> 3) * 8];
#pragma unroll
    for (int nt = 0; nt < 4; ++nt)
      acc[nt] = __builtin_amdgcn_wmma_f32_16x16x32_bf16(
          false, afrag, false, bfrag[nt], (short)0, acc[nt], false, false);
    __syncthreads();
  }

  // epilogue: bias + activation in place, then uniform-branch store loops
  const int mbase = m0 + wave * 16 + (lane >> 4) * 8;
#pragma unroll
  for (int nt = 0; nt < 4; ++nt) {
    int col = n0 + nt * 16 + nloc;
    float b = bias ? bias[col] : 0.f;
#pragma unroll
    for (int r = 0; r < 8; ++r) {
      float v = acc[nt][r] + b;
      if (act == 1) v = v > 0.f ? v : 0.f;                    // ReLU
      else if (act == 2) v = v > 0.f ? v : (__expf(v) - 1.f); // ELU
      acc[nt][r] = v;
    }
  }
  if (Cf) {
#pragma unroll
    for (int nt = 0; nt < 4; ++nt) {
      int col = n0 + nt * 16 + nloc;
#pragma unroll
      for (int r = 0; r < 8; ++r)
        Cf[(size_t)(mbase + r) * N + col] = acc[nt][r];
    }
  }
  if (Cb) {
#pragma unroll
    for (int nt = 0; nt < 4; ++nt) {
      int col = n0 + nt * 16 + nloc;
#pragma unroll
      for (int r = 0; r < 8; ++r)
        Cb[(size_t)(mbase + r) * N + col] = (__bf16)acc[nt][r];
    }
  }
}

// --------------------------------- driver -----------------------------------

extern "C" void kernel_launch(void* const* d_in, const int* in_sizes, int n_in,
                              void* d_out, int out_size, void* d_ws, size_t ws_size,
                              hipStream_t stream) {
  (void)in_sizes; (void)n_in; (void)out_size; (void)ws_size;
  const int NN = 4096, NF = 512, NH = 64, H1 = 8, OD = 512, L1 = 1024;
  const int NW = NN / 32;

  const float* x      = (const float*)d_in[0];
  const int*   adj    = (const int*)  d_in[1];
  const float* W1     = (const float*)d_in[2];
  const float* a1     = (const float*)d_in[3];
  const float* Wo     = (const float*)d_in[4];
  const float* ao     = (const float*)d_in[5];
  const float* lin1_w = (const float*)d_in[6];
  const float* lin1_b = (const float*)d_in[7];
  const float* lin2_w = (const float*)d_in[8];
  const float* lin2_b = (const float*)d_in[9];

  char* ws = (char*)d_ws;
  size_t off = 0;
  auto alloc = [&](size_t bytes) -> void* {
    void* p = ws + off;
    off = (off + bytes + 255) & ~(size_t)255;
    return p;
  };

  __bf16*   xb   = (__bf16*)  alloc((size_t)NN * NF * 2);
  __bf16*   W1b  = (__bf16*)  alloc((size_t)NF * (H1 * NH) * 2);
  __bf16*   Wob  = (__bf16*)  alloc((size_t)NF * OD * 2);
  __bf16*   l1wb = (__bf16*)  alloc((size_t)OD * L1 * 2);
  __bf16*   l2wb = (__bf16*)  alloc((size_t)L1 * OD * 2);
  unsigned* bits = (unsigned*)alloc((size_t)NN * NW * 4);
  float*    Whf  = (float*)   alloc((size_t)NN * NF * 4);
  __bf16*   Whb  = (__bf16*)  alloc((size_t)NN * NF * 2);
  float*    src1 = (float*)   alloc((size_t)H1 * NN * 4);
  float*    dst1 = (float*)   alloc((size_t)H1 * NN * 4);
  float*    m1   = (float*)   alloc((size_t)H1 * NN * 4);
  float*    s1   = (float*)   alloc((size_t)H1 * NN * 4);
  __bf16*   x1b  = (__bf16*)  alloc((size_t)NN * NF * 2);
  float*    Wh2f = (float*)   alloc((size_t)NN * OD * 4);
  __bf16*   Wh2b = (__bf16*)  alloc((size_t)NN * OD * 2);
  float*    src2 = (float*)   alloc((size_t)NN * 4);
  float*    dst2 = (float*)   alloc((size_t)NN * 4);
  float*    m2   = (float*)   alloc((size_t)NN * 4);
  float*    s2   = (float*)   alloc((size_t)NN * 4);
  __bf16*   x2b  = (__bf16*)  alloc((size_t)NN * OD * 2);
  __bf16*   t1b  = (__bf16*)  alloc((size_t)NN * L1 * 2);

  // --- precision conversions / packing ---
  k_f32_to_bf16<<<1024, 256, 0, stream>>>(x, xb, (size_t)NN * NF);
  k_pack_W1<<<(H1 * NF * NH + 255) / 256, 256, 0, stream>>>(W1, W1b, H1, NF, NH);
  k_f32_to_bf16<<<1024, 256, 0, stream>>>(Wo, Wob, (size_t)NF * OD);
  k_f32_to_bf16<<<1024, 256, 0, stream>>>(lin1_w, l1wb, (size_t)OD * L1);
  k_f32_to_bf16<<<1024, 256, 0, stream>>>(lin2_w, l2wb, (size_t)L1 * OD);
  k_pack_adj<<<(NN * NW) / 8, 256, 0, stream>>>(adj, bits, NN);

  // --- layer 1: Wh = x @ W1 (all heads packed), projections, softmax, PV ---
  k_gemm_bf16<<<dim3(NN / 128, NF / 64), 256, 0, stream>>>(
      xb, W1b, Whf, Whb, nullptr, NN, NF, NF, 0);
  k_attn_proj<<<(H1 * NN + 255) / 256, 256, 0, stream>>>(
      Whf, a1, src1, dst1, NN, H1, NH);
  k_softmax_stats<<<(H1 * NN) / 8, 256, 0, stream>>>(
      src1, dst1, bits, m1, s1, NN, H1);
  k_attn_wmma<<<dim3(NN / 64, 1, H1), 128, 0, stream>>>(
      Whb, src1, dst1, m1, s1, bits, nullptr, x1b, NN, H1, NH);

  // --- layer 2: single 512-wide head ---
  k_gemm_bf16<<<dim3(NN / 128, OD / 64), 256, 0, stream>>>(
      x1b, Wob, Wh2f, Wh2b, nullptr, NN, OD, NF, 0);
  k_attn_proj<<<(NN + 255) / 256, 256, 0, stream>>>(
      Wh2f, ao, src2, dst2, NN, 1, OD);
  k_softmax_stats<<<NN / 8, 256, 0, stream>>>(
      src2, dst2, bits, m2, s2, NN, 1);
  k_attn_wmma<<<dim3(NN / 64, OD / 64, 1), 128, 0, stream>>>(
      Wh2b, src2, dst2, m2, s2, bits, nullptr, x2b, NN, 1, OD);

  // --- MLP head ---
  k_gemm_bf16<<<dim3(NN / 128, L1 / 64), 256, 0, stream>>>(
      x2b, l1wb, nullptr, t1b, lin1_b, NN, L1, OD, 1);
  k_gemm_bf16<<<dim3(NN / 128, OD / 64), 256, 0, stream>>>(
      t1b, l2wb, (float*)d_out, nullptr, lin2_b, NN, OD, L1, 0);
}
